// CombinedHiddenGCAEDecoder_16286515987227
// MI455X (gfx1250) — matrix operand order repaired and checked by reading, not compiled, not run on hardware
//
#include <hip/hip_runtime.h>
#include <stdint.h>

#define N_NODES 10000
#define N_EDGES 160000
#define IN_DIM  256
#define HID_DIM 512
#define OUT_DIM 256

typedef __attribute__((ext_vector_type(16))) _Float16 v16h;
typedef __attribute__((ext_vector_type(8)))  float    v8f;

// ---------------------------------------------------------------------------
// Native CDNA5 fp32 global atomic add (no-return form -> STOREcnt, no CAS loop)
// ---------------------------------------------------------------------------
__device__ __forceinline__ void atomic_add_f32_nr(float* p, float v) {
  asm volatile("global_atomic_add_f32 %0, %1, off" :: "v"(p), "v"(v) : "memory");
}

// ---------------------------------------------------------------------------
// Degree / normalization
// ---------------------------------------------------------------------------
__global__ __launch_bounds__(256) void deg_kernel(const int* __restrict__ dst,
                                                  unsigned* __restrict__ deg, int E) {
  int e = blockIdx.x * blockDim.x + threadIdx.x;
  if (e < E) atomicAdd(&deg[dst[e]], 1u);
}

__global__ __launch_bounds__(256) void dinv_kernel(const unsigned* __restrict__ deg,
                                                   float* __restrict__ dinv, int N) {
  int i = blockIdx.x * blockDim.x + threadIdx.x;
  if (i < N) dinv[i] = rsqrtf((float)deg[i] + 1.0f);
}

// Per-edge symmetric norm, shared by all three layers
__global__ __launch_bounds__(256) void norm_kernel(const int* __restrict__ src,
                                                   const int* __restrict__ dst,
                                                   const float* __restrict__ dinv,
                                                   float* __restrict__ norm, int E) {
  int e = blockIdx.x * blockDim.x + threadIdx.x;
  if (e < E) norm[e] = dinv[src[e]] * dinv[dst[e]];
}

// ---------------------------------------------------------------------------
// f32 -> packed f16 conversion (activation input for the first GEMM)
// ---------------------------------------------------------------------------
__global__ __launch_bounds__(256) void cvt_f32_f16_kernel(const float* __restrict__ in,
                                                          _Float16* __restrict__ out,
                                                          long n) {
  long i = (long)blockIdx.x * blockDim.x + threadIdx.x;
  if (i < n) out[i] = (_Float16)in[i];
}

// ---------------------------------------------------------------------------
// Pack W [K, FOUT] (f32 row-major) into WMMA B-fragment order (f16 pairs),
// with each lane's 8 fragment dwords CONTIGUOUS so the GEMM fetches B with
// two global_load_b128 per fragment:
//   t = ((kt*(FOUT/16) + ntile)*32 + lane)*8 + v
//   n = ntile*16 + (lane&15), k = kt*32 + (lane>>4)*16 + 2v -> pack(W[k,n],W[k+1,n])
// ---------------------------------------------------------------------------
__global__ __launch_bounds__(256) void pack_w_kernel(const float* __restrict__ W,
                                                     uint32_t* __restrict__ Bp,
                                                     int K, int FOUT) {
  int t = blockIdx.x * blockDim.x + threadIdx.x;
  int total = (K * FOUT) >> 1;
  if (t >= total) return;
  int v      = t & 7;
  int lane   = (t >> 3) & 31;
  int rest   = t >> 8;
  int ntiles = FOUT >> 4;
  int ntile  = rest % ntiles;
  int kt     = rest / ntiles;
  int n = ntile * 16 + (lane & 15);
  int k = kt * 32 + (lane >> 4) * 16 + 2 * v;
  union { _Float16 h[2]; uint32_t u; } cv;
  cv.h[0] = (_Float16)W[(size_t)k       * FOUT + n];
  cv.h[1] = (_Float16)W[(size_t)(k + 1) * FOUT + n];
  Bp[t] = cv.u;
}

// ---------------------------------------------------------------------------
// WMMA GEMM: C[N, FOUT] = A[N, K] * B[K, FOUT]
//   A2: row-major f16, viewed as uint32 pairs [N, K/2]  (2x b128 loads/frag)
//   Bp: pre-packed B fragments, lane-contiguous        (2x b128 loads/frag)
// One wave computes one 16x16 C tile; 8 waves/block -> 16 rows x 128 cols.
// Grid: (N/16, FOUT/128). EXEC is all-ones (exact grid), as WMMA requires.
// ---------------------------------------------------------------------------
template <int K, int FOUT>
__global__ __launch_bounds__(256) void gemm_wmma_kernel(const uint32_t* __restrict__ A2,
                                                        const uint32_t* __restrict__ Bp,
                                                        float* __restrict__ C) {
  constexpr int K2      = K / 2;
  constexpr int BSTRIDE = (FOUT / 16) * 8 * 32;  // dwords per kt step of B

  const int lane    = threadIdx.x & 31;
  const int wave    = threadIdx.x >> 5;
  const int m0      = blockIdx.x * 16;
  const int ntile   = blockIdx.y * 8 + wave;
  const int half_id = lane >> 4;
  const int mloc    = lane & 15;

  const uint32_t* arow  = A2 + (size_t)(m0 + mloc) * K2;
  const uint32_t* bbase = Bp + (size_t)ntile * 256 + lane * 8;

  v8f c = {};
#pragma unroll
  for (int kt = 0; kt < K / 32; ++kt) {
    union { uint32_t u[8]; v16h h; } a, b;
    const uint32_t* ak = arow + kt * 16 + half_id * 8;
#pragma unroll
    for (int v = 0; v < 8; ++v) a.u[v] = ak[v];  // 2x b128
    const uint32_t* bk = bbase + (size_t)kt * BSTRIDE;
#pragma unroll
    for (int v = 0; v < 8; ++v) b.u[v] = bk[v];                       // 2x b128
    c = __builtin_amdgcn_wmma_f32_16x16x32_f16(false, a.h, false, b.h,
                                               (short)0, c, false, false);
  }

  const int n = ntile * 16 + mloc;
#pragma unroll
  for (int r = 0; r < 8; ++r)
    C[(size_t)(m0 + r + 8 * half_id) * FOUT + n] = c[r];
}

// ---------------------------------------------------------------------------
// Edge scatter: agg[dst] += h[src] * norm[e]
// Consecutive lanes handle consecutive features of one edge (coalesced gather).
// Templated on F so the index decomposition is shifts, not division.
// ---------------------------------------------------------------------------
template <int F>
__global__ __launch_bounds__(256) void scatter_kernel(const int* __restrict__ src,
                                                      const int* __restrict__ dst,
                                                      const float* __restrict__ norm,
                                                      const float* __restrict__ h,
                                                      float* __restrict__ agg) {
  constexpr int F4 = F / 4;
  long tid = (long)blockIdx.x * blockDim.x + threadIdx.x;
  int e = (int)(tid / F4);                // F4 is a power of two -> shift
  if (e >= N_EDGES) return;
  int c = (int)(tid & (F4 - 1)) * 4;
  int s = src[e], d = dst[e];
  float nrm = norm[e];
  const float4 v = *(const float4*)(h + (size_t)s * F + c);
  float* a = agg + (size_t)d * F + c;
  atomic_add_f32_nr(a + 0, v.x * nrm);
  atomic_add_f32_nr(a + 1, v.y * nrm);
  atomic_add_f32_nr(a + 2, v.z * nrm);
  atomic_add_f32_nr(a + 3, v.w * nrm);
}

// ---------------------------------------------------------------------------
// Finalize: agg + h*dinv^2 + b, then tanh -> f16 (hidden layers)
// ---------------------------------------------------------------------------
__global__ __launch_bounds__(256) void finalize_tanh_kernel(const float* __restrict__ agg,
                                                            const float* __restrict__ h,
                                                            const float* __restrict__ dinv,
                                                            const float* __restrict__ b,
                                                            _Float16* __restrict__ out,
                                                            int F, long total) {
  long i = (long)blockIdx.x * blockDim.x + threadIdx.x;
  if (i >= total) return;
  int row = (int)(i / F);
  int col = (int)(i - (long)row * F);
  float di = dinv[row];
  float v = agg[i] + h[i] * di * di + b[col];
  out[i] = (_Float16)tanhf(v);
}

// Final layer: f32 output, no tanh
__global__ __launch_bounds__(256) void finalize_out_kernel(const float* __restrict__ agg,
                                                           const float* __restrict__ h,
                                                           const float* __restrict__ dinv,
                                                           const float* __restrict__ b,
                                                           float* __restrict__ out,
                                                           int F, long total) {
  long i = (long)blockIdx.x * blockDim.x + threadIdx.x;
  if (i >= total) return;
  int row = (int)(i / F);
  int col = (int)(i - (long)row * F);
  float di = dinv[row];
  out[i] = agg[i] + h[i] * di * di + b[col];
}

// ---------------------------------------------------------------------------
// Host launch
// ---------------------------------------------------------------------------
extern "C" void kernel_launch(void* const* d_in, const int* in_sizes, int n_in,
                              void* d_out, int out_size, void* d_ws, size_t ws_size,
                              hipStream_t stream) {
  (void)in_sizes; (void)n_in; (void)out_size; (void)ws_size;
  const float* x  = (const float*)d_in[0];
  const float* W1 = (const float*)d_in[1];
  const float* b1 = (const float*)d_in[2];
  const float* W2 = (const float*)d_in[3];
  const float* b2 = (const float*)d_in[4];
  const float* W3 = (const float*)d_in[5];
  const float* b3 = (const float*)d_in[6];
  const int*   ei = (const int*)d_in[7];
  const int* src = ei;
  const int* dst = ei + N_EDGES;

  char* ws = (char*)d_ws;
  size_t off = 0;
  auto alloc = [&](size_t bytes) {
    char* p = ws + off;
    off = (off + bytes + 255) & ~(size_t)255;
    return p;
  };
  unsigned* deg  = (unsigned*)alloc((size_t)N_NODES * 4);
  float*    dinv = (float*)   alloc((size_t)N_NODES * 4);
  float*    nrm  = (float*)   alloc((size_t)N_EDGES * 4);
  _Float16* actA = (_Float16*)alloc((size_t)N_NODES * HID_DIM * 2);  // ping
  _Float16* actB = (_Float16*)alloc((size_t)N_NODES * HID_DIM * 2);  // pong
  float*    h    = (float*)   alloc((size_t)N_NODES * HID_DIM * 4);
  float*    agg  = (float*)   alloc((size_t)N_NODES * HID_DIM * 4);
  uint32_t* Wp1  = (uint32_t*)alloc((size_t)IN_DIM  * HID_DIM * 2);
  uint32_t* Wp2  = (uint32_t*)alloc((size_t)HID_DIM * HID_DIM * 2);
  uint32_t* Wp3  = (uint32_t*)alloc((size_t)HID_DIM * OUT_DIM * 2);

  const int B = 256;
  auto blocks = [](long n, int b) { return (int)((n + b - 1) / b); };

  // --- degrees / dinv / per-edge norm (shared by all layers) ---
  hipMemsetAsync(deg, 0, (size_t)N_NODES * 4, stream);
  deg_kernel<<<blocks(N_EDGES, B), B, 0, stream>>>(dst, deg, N_EDGES);
  dinv_kernel<<<blocks(N_NODES, B), B, 0, stream>>>(deg, dinv, N_NODES);
  norm_kernel<<<blocks(N_EDGES, B), B, 0, stream>>>(src, dst, dinv, nrm, N_EDGES);

  // --- one-time precision/layout transforms ---
  cvt_f32_f16_kernel<<<blocks((long)N_NODES * IN_DIM, B), B, 0, stream>>>(
      x, actA, (long)N_NODES * IN_DIM);
  pack_w_kernel<<<blocks((long)IN_DIM  * HID_DIM / 2, B), B, 0, stream>>>(W1, Wp1, IN_DIM,  HID_DIM);
  pack_w_kernel<<<blocks((long)HID_DIM * HID_DIM / 2, B), B, 0, stream>>>(W2, Wp2, HID_DIM, HID_DIM);
  pack_w_kernel<<<blocks((long)HID_DIM * OUT_DIM / 2, B), B, 0, stream>>>(W3, Wp3, HID_DIM, OUT_DIM);

  const int MT = N_NODES / 16;  // 625 M-tiles

  // ---------------- Layer 1: [N,256] x [256,512] ----------------
  gemm_wmma_kernel<IN_DIM, HID_DIM><<<dim3(MT, HID_DIM / 128), B, 0, stream>>>(
      (const uint32_t*)actA, Wp1, h);
  hipMemsetAsync(agg, 0, (size_t)N_NODES * HID_DIM * 4, stream);
  scatter_kernel<HID_DIM><<<blocks((long)N_EDGES * (HID_DIM / 4), B), B, 0, stream>>>(
      src, dst, nrm, h, agg);
  finalize_tanh_kernel<<<blocks((long)N_NODES * HID_DIM, B), B, 0, stream>>>(
      agg, h, dinv, b1, actB, HID_DIM, (long)N_NODES * HID_DIM);

  // ---------------- Layer 2: [N,512] x [512,512] ----------------
  gemm_wmma_kernel<HID_DIM, HID_DIM><<<dim3(MT, HID_DIM / 128), B, 0, stream>>>(
      (const uint32_t*)actB, Wp2, h);
  hipMemsetAsync(agg, 0, (size_t)N_NODES * HID_DIM * 4, stream);
  scatter_kernel<HID_DIM><<<blocks((long)N_EDGES * (HID_DIM / 4), B), B, 0, stream>>>(
      src, dst, nrm, h, agg);
  finalize_tanh_kernel<<<blocks((long)N_NODES * HID_DIM, B), B, 0, stream>>>(
      agg, h, dinv, b2, actA, HID_DIM, (long)N_NODES * HID_DIM);

  // ---------------- Layer 3: [N,512] x [512,256] ----------------
  gemm_wmma_kernel<HID_DIM, OUT_DIM><<<dim3(MT, OUT_DIM / 128), B, 0, stream>>>(
      (const uint32_t*)actA, Wp3, h);
  hipMemsetAsync(agg, 0, (size_t)N_NODES * OUT_DIM * 4, stream);
  scatter_kernel<OUT_DIM><<<blocks((long)N_EDGES * (OUT_DIM / 4), B), B, 0, stream>>>(
      src, dst, nrm, h, agg);
  finalize_out_kernel<<<blocks((long)N_NODES * OUT_DIM, B), B, 0, stream>>>(
      agg, h, dinv, b3, (float*)d_out, OUT_DIM, (long)N_NODES * OUT_DIM);
}
